// MHACausal_42167988912450
// MI455X (gfx1250) — matrix operand order repaired and checked by reading, not compiled, not run on hardware
//
#include <hip/hip_runtime.h>

// Causal MHA forward for MI455X (gfx1250, wave32).
// Compute-bound (~206 GFLOP vs ~8.6us of HBM traffic); all matmuls run as
// v_wmma_f32_16x16x32_bf16 with fp32 accumulation. bf16 staging copies of
// every operand (~112 MB) fit entirely in the 192 MB L2, so fragment loads go
// straight to global (L2-resident) with b128 accesses shaped to the ISA's
// WMMA A/B VGPR layouts. GEMM waves use 64x64 register tiles (16 WMMA per
// 16 b128 loads = 32 FLOP per L2 byte) so L2 bandwidth is not the limiter.
// V is staged transposed [B,H,HD,S] so the PV B-matrix loads contiguously.

#define B_   2
#define S_   2048
#define D_   2048
#define H_   16
#define HD_  128
#define M_   (B_ * S_)   // 4096 rows for the x-side GEMMs

typedef __attribute__((ext_vector_type(16))) __bf16 bf16x16;
typedef __attribute__((ext_vector_type(8)))  float  f32x8;

struct alignas(16) I4 { int x, y, z, w; };
struct alignas(16) F4s { float x, y, z, w; };
struct alignas(8)  U4s { unsigned short x, y, z, w; };
union FragAB { bf16x16 v; I4 q[2]; };

__device__ __forceinline__ unsigned short f2bf_bits(float f) {
  union { float f; unsigned int u; } a; a.f = f;
  unsigned int u = a.u;
  return (unsigned short)((u + 0x7FFFu + ((u >> 16) & 1u)) >> 16);  // RNE
}
__device__ __forceinline__ __bf16 f2bf(float f) {
  union { unsigned short s; __bf16 b; } o; o.s = f2bf_bits(f);
  return o.b;
}

// A-fragment: 16x32 bf16, row-major source (lda in elements).
// Lane L holds row L%16; K chunks [8*(L/16)..+7] and [16+8*(L/16)..+7].
__device__ __forceinline__ bf16x16 load_a_frag(const __bf16* base, int lda, int lane) {
  int r = lane & 15, hf = lane >> 4;
  const __bf16* p = base + (size_t)r * lda + hf * 8;
  FragAB f;
  f.q[0] = *(const I4*)(p);
  f.q[1] = *(const I4*)(p + 16);
  return f.v;
}
// B-fragment: 32x16 bf16 (KxN) whose column n is row n of a row-major [N,K]
// source. Lane L holds column L%16; K chunks [8*(L/16)..+7] and [16+8*(L/16)..+7].
__device__ __forceinline__ bf16x16 load_b_frag(const __bf16* base, int ldb, int lane) {
  int c = lane & 15, hf = lane >> 4;
  const __bf16* p = base + (size_t)c * ldb + hf * 8;
  FragAB f;
  f.q[0] = *(const I4*)(p);
  f.q[1] = *(const I4*)(p + 16);
  return f.v;
}
__device__ __forceinline__ f32x8 wmma_bf16(bf16x16 a, bf16x16 b, f32x8 c) {
  return __builtin_amdgcn_wmma_f32_16x16x32_bf16(false, a, false, b, (short)0, c,
                                                 false, false);
}

// ---------------------------------------------------------------- fp32->bf16
__global__ __launch_bounds__(256) void cvt_f32_to_bf16(const float* __restrict__ src,
                                                       unsigned short* __restrict__ dst,
                                                       int n4) {
  int i = blockIdx.x * 256 + threadIdx.x;
  if (i >= n4) return;
  F4s f = ((const F4s*)src)[i];
  U4s o;
  o.x = f2bf_bits(f.x); o.y = f2bf_bits(f.y);
  o.z = f2bf_bits(f.z); o.w = f2bf_bits(f.w);
  ((U4s*)dst)[i] = o;
}

// ----------------------------------------------------- y = A @ W^T (bf16 WMMA)
// A: [M_, D_] bf16 row-major, W: [D_, D_] bf16 row-major (torch Linear: K rows).
// Each wave owns a 64x64 output tile: 4x4 WMMA frags, 16 WMMAs per 32-K step
// against 16 b128 fragment loads. 8 waves/block arranged 2(m) x 4(n) ->
// block tile 128x256. Grid (M_/128, D_/256).
// MODE is a template parameter so each instantiation has one straight-line
// epilogue; per-fragment bases are pure scalar math (h and b are uniform
// within a fragment) plus one precomputed per-lane offset, and row stores use
// constant immediate offsets. This keeps the 128 accumulator VGPRs resident
// (no scratch spill at loop exit).
// MODE 0 (Q): outb <- bf16 split-heads [B,H,S,HD]
// MODE 1 (K): outb <- bf16 [B,H,S,HD], outf <- fp32 [B,H,S,HD]
// MODE 2 (V): outb <- bf16 TRANSPOSED [B,H,HD,S], outf <- fp32 [B,H,S,HD]
// MODE 3 (O): outf <- fp32 [M_, D_]
template <int MODE>
__global__ __launch_bounds__(256) void gemm_bf16(const __bf16* __restrict__ A,
                                                 const __bf16* __restrict__ W,
                                                 __bf16* __restrict__ outb,
                                                 float* __restrict__ outf) {
  const int lane = threadIdx.x & 31;
  const int wave = threadIdx.x >> 5;
  const int m0 = blockIdx.x * 128 + (wave >> 2) * 64;
  const int n0 = blockIdx.y * 256 + (wave & 3) * 64;

  f32x8 acc[4][4] = {};
  const __bf16* ap[4];
  const __bf16* wp[4];
  for (int i = 0; i < 4; ++i) {
    ap[i] = A + (size_t)(m0 + 16 * i) * D_;
    wp[i] = W + (size_t)(n0 + 16 * i) * D_;
  }

  for (int k = 0; k < D_; k += 32) {
    __builtin_prefetch(ap[0] + k + 256, 0, 1);   // global_prefetch next K tiles
    __builtin_prefetch(wp[0] + k + 256, 0, 1);
    bf16x16 af[4], wf[4];
    for (int i = 0; i < 4; ++i) af[i] = load_a_frag(ap[i] + k, D_, lane);
    for (int j = 0; j < 4; ++j) wf[j] = load_b_frag(wp[j] + k, D_, lane);
    for (int i = 0; i < 4; ++i)
      for (int j = 0; j < 4; ++j)
        acc[i][j] = wmma_bf16(af[i], wf[j], acc[i][j]);
  }

  // C layout: VGPR r, lane L -> row r + 8*(L/16), col L%16.
  const int cc = lane & 15, hf = lane >> 4;
  // Per-lane offset vectors (the ONLY vector part of each store address):
  const int voff_f  = hf * 8 * D_ + cc;    // [M,D] rows, stride D_
  const int voff_hs = hf * 8 * HD_ + cc;   // [B,H,S,HD] rows, stride HD_
  const int voff_vt = cc * S_ + hf * 8;    // V-transposed cols, stride S_
  for (int ti = 0; ti < 4; ++ti) {
    const int mrow0 = m0 + ti * 16;                 // uniform
    const int b = mrow0 >> 11;                      // uniform within fragment
    const int sr = mrow0 & (S_ - 1);                // uniform
    for (int tj = 0; tj < 4; ++tj) {
      f32x8 c = acc[ti][tj];
      const int nc0 = n0 + tj * 16;                 // uniform
      if (MODE == 3) {
        float* p = outf + (size_t)mrow0 * D_ + nc0 + voff_f;
        for (int r = 0; r < 8; ++r) p[r * D_] = c[r];
      } else {
        const int h = nc0 >> 7, hd0 = nc0 & (HD_ - 1);   // uniform
        const size_t bh = (size_t)(b * H_ + h);
        if (MODE == 2) {
          __bf16* pb = outb + (bh * HD_ + hd0) * S_ + sr + voff_vt;
          float*  pf = outf + (bh * S_ + sr) * HD_ + hd0 + voff_hs;
          for (int r = 0; r < 8; ++r) {
            pb[r] = f2bf(c[r]);
            pf[r * HD_] = c[r];
          }
        } else if (MODE == 1) {
          __bf16* pb = outb + (bh * S_ + sr) * HD_ + hd0 + voff_hs;
          float*  pf = outf + (bh * S_ + sr) * HD_ + hd0 + voff_hs;
          for (int r = 0; r < 8; ++r) {
            pb[r * HD_] = f2bf(c[r]);
            pf[r * HD_] = c[r];
          }
        } else {
          __bf16* pb = outb + (bh * S_ + sr) * HD_ + hd0 + voff_hs;
          for (int r = 0; r < 8; ++r) pb[r * HD_] = f2bf(c[r]);
        }
      }
    }
  }
}

// ------------------------------------------------------ causal flash attention
// 1 wave per 16-row q-tile per (b,h); 32-key chunks; streaming softmax.
// qb,kb: bf16 [B,H,S,HD]; vtb: bf16 [B,H,HD,S]; ab out: bf16 merged [B,S,D].
#define LDS_STRIDE 40   // 16x32 P tile, padded rows (80B, 16B-aligned)
__global__ __launch_bounds__(128) void attn_causal(const __bf16* __restrict__ qb,
                                                   const __bf16* __restrict__ kb,
                                                   const __bf16* __restrict__ vtb,
                                                   __bf16* __restrict__ ab) {
  __shared__ __bf16 plds[4][16 * LDS_STRIDE];   // per-wave private region
  const int lane = threadIdx.x & 31;
  const int wave = threadIdx.x >> 5;
  const int q0 = (blockIdx.x * 4 + wave) * 16;
  const int bh = blockIdx.y;               // b*16 + h
  const int b = bh >> 4, h = bh & 15;
  const int cc = lane & 15, hf = lane >> 4;

  const __bf16* qbase = qb + (size_t)bh * S_ * HD_;
  const __bf16* kbase = kb + (size_t)bh * S_ * HD_;
  const __bf16* vbase = vtb + (size_t)bh * HD_ * S_;
  __bf16* pl = &plds[wave][0];

  bf16x16 qf[4];
  for (int t = 0; t < 4; ++t)
    qf[t] = load_a_frag(qbase + (size_t)q0 * HD_ + t * 32, HD_, lane);

  f32x8 o[8];
  float m[8], l[8];
  for (int n = 0; n < 8; ++n) o[n] = (f32x8){};
  for (int r = 0; r < 8; ++r) { m[r] = -3.402823466e38f; l[r] = 0.0f; }

  const float scale = 0.08838834764831845f;   // 1/sqrt(128)
  const int kend = q0 + 16;                   // keys j <= i < q0+16

  for (int kc = 0; kc < kend; kc += 32) {
    // ---- S = Q @ K^T for key columns [kc, kc+32)
    f32x8 s0 = {}, s1 = {};
    for (int t = 0; t < 4; ++t) {
      bf16x16 kf0 = load_b_frag(kbase + (size_t)kc * HD_ + t * 32, HD_, lane);
      bf16x16 kf1 = load_b_frag(kbase + (size_t)(kc + 16) * HD_ + t * 32, HD_, lane);
      s0 = wmma_bf16(qf[t], kf0, s0);
      s1 = wmma_bf16(qf[t], kf1, s1);
    }
    __builtin_prefetch(kbase + (size_t)(kc + 32) * HD_, 0, 1);

    // ---- scale + causal mask + streaming softmax (rows live in 16-lane halves)
    const int j0 = kc + cc, j1 = j0 + 16;
    for (int r = 0; r < 8; ++r) {
      int irow = q0 + hf * 8 + r;
      float v0 = (j0 <= irow) ? s0[r] * scale : -3.402823466e38f;
      float v1 = (j1 <= irow) ? s1[r] * scale : -3.402823466e38f;
      float mx = fmaxf(v0, v1);
      mx = fmaxf(mx, __shfl_xor(mx, 1, 32));
      mx = fmaxf(mx, __shfl_xor(mx, 2, 32));
      mx = fmaxf(mx, __shfl_xor(mx, 4, 32));
      mx = fmaxf(mx, __shfl_xor(mx, 8, 32));
      float mn = fmaxf(m[r], mx);
      float alpha = __expf(m[r] - mn);
      m[r] = mn;
      float p0 = __expf(v0 - mn);
      float p1 = __expf(v1 - mn);
      s0[r] = p0; s1[r] = p1;
      float rs = p0 + p1;
      rs += __shfl_xor(rs, 1, 32);
      rs += __shfl_xor(rs, 2, 32);
      rs += __shfl_xor(rs, 4, 32);
      rs += __shfl_xor(rs, 8, 32);
      l[r] = l[r] * alpha + rs;
      for (int n = 0; n < 8; ++n) o[n][r] *= alpha;
    }

    // ---- re-shape P: C layout -> A layout via per-wave LDS tile (bf16)
    for (int r = 0; r < 8; ++r) {
      int il = hf * 8 + r;
      pl[il * LDS_STRIDE + cc] = f2bf(s0[r]);
      pl[il * LDS_STRIDE + cc + 16] = f2bf(s1[r]);
    }
    asm volatile("" ::: "memory");   // same-wave LDS ops are in-order in HW
    FragAB pf;
    pf.q[0] = *(const I4*)(pl + cc * LDS_STRIDE + hf * 8);
    pf.q[1] = *(const I4*)(pl + cc * LDS_STRIDE + 16 + hf * 8);
    asm volatile("" ::: "memory");

    // ---- O += P @ V  (V transposed: column hd -> contiguous key run)
    for (int n = 0; n < 8; ++n) {
      bf16x16 vf = load_b_frag(vbase + (size_t)(16 * n) * S_ + kc, S_, lane);
      o[n] = wmma_bf16(pf.v, vf, o[n]);
    }
  }

  // ---- normalize + write merged [B,S,D] as bf16 for the Wo GEMM
  for (int r = 0; r < 8; ++r) {
    float inv = __builtin_amdgcn_rcpf(l[r]);
    int srow = q0 + hf * 8 + r;
    __bf16* orow = ab + ((size_t)b * S_ + srow) * D_ + h * HD_ + cc;
    for (int n = 0; n < 8; ++n)
      orow[16 * n] = f2bf(o[n][r] * inv);
  }
}

// ------------------------------------------------------------------- launcher
extern "C" void kernel_launch(void* const* d_in, const int* in_sizes, int n_in,
                              void* d_out, int out_size, void* d_ws, size_t ws_size,
                              hipStream_t stream) {
  const float* x  = (const float*)d_in[0];
  const float* Wq = (const float*)d_in[1];
  const float* Wk = (const float*)d_in[2];
  const float* Wv = (const float*)d_in[3];
  const float* Wo = (const float*)d_in[4];
  float* out = (float*)d_out;                       // [B,S,D] fp32
  float* out_k = out + (size_t)B_ * S_ * D_;        // [B,H,S,HD] fp32
  float* out_v = out_k + (size_t)B_ * S_ * D_;      // [B,H,S,HD] fp32

  // bf16 staging in workspace (~112 MB; fits L2)
  char* ws = (char*)d_ws;
  __bf16* xb  = (__bf16*)(ws);                      // [M_, D_]
  __bf16* wqb = (__bf16*)(ws + (size_t)16777216);
  __bf16* wkb = (__bf16*)(ws + (size_t)25165824);
  __bf16* wvb = (__bf16*)(ws + (size_t)33554432);
  __bf16* wob = (__bf16*)(ws + (size_t)41943040);
  __bf16* qbb = (__bf16*)(ws + (size_t)50331648);   // [B,H,S,HD]
  __bf16* kbb = (__bf16*)(ws + (size_t)67108864);   // [B,H,S,HD]
  __bf16* vtb = (__bf16*)(ws + (size_t)83886080);   // [B,H,HD,S]
  __bf16* abb = (__bf16*)(ws + (size_t)100663296);  // [B,S,D] merged attn out

  const int nx4 = (B_ * S_ * D_) / 4;   // 2097152
  const int nw4 = (D_ * D_) / 4;        // 1048576
  cvt_f32_to_bf16<<<nx4 / 256, 256, 0, stream>>>(x,  (unsigned short*)xb,  nx4);
  cvt_f32_to_bf16<<<nw4 / 256, 256, 0, stream>>>(Wq, (unsigned short*)wqb, nw4);
  cvt_f32_to_bf16<<<nw4 / 256, 256, 0, stream>>>(Wk, (unsigned short*)wkb, nw4);
  cvt_f32_to_bf16<<<nw4 / 256, 256, 0, stream>>>(Wv, (unsigned short*)wvb, nw4);
  cvt_f32_to_bf16<<<nw4 / 256, 256, 0, stream>>>(Wo, (unsigned short*)wob, nw4);

  dim3 ggrid(M_ / 128, D_ / 256);       // 32 x 8, 8 waves/block (2m x 4n)
  gemm_bf16<0><<<ggrid, 256, 0, stream>>>(xb, wqb, qbb, (float*)nullptr);
  gemm_bf16<1><<<ggrid, 256, 0, stream>>>(xb, wkb, kbb, out_k);
  gemm_bf16<2><<<ggrid, 256, 0, stream>>>(xb, wvb, vtb, out_v);

  dim3 agrid(S_ / 16 / 4, B_ * H_);     // 32 x 32, 4 waves/block
  attn_causal<<<agrid, 128, 0, stream>>>(qbb, kbb, vtb, abb);

  gemm_bf16<3><<<ggrid, 256, 0, stream>>>(abb, wob, (__bf16*)nullptr, out);
}
